// TransformerEncoder_54580444397764
// MI455X (gfx1250) — matrix operand order repaired and checked by reading, not compiled, not run on hardware
//
#include <hip/hip_runtime.h>
#include <hip/hip_bf16.h>
#include <math.h>

// ---------------------------------------------------------------------------
// Transformer encoder for MI455X (gfx1250).
// - GEMMs (99.7% of FLOPs): V_WMMA_F32_16X16X32_BF16, fp32 accumulate.
// - Weights converted fp32->bf16 once per layer; LN/attention/FFN1 emit bf16
//   activations, so the GEMM K-loop is conversion-free.
// - Tiles staged to LDS with GLOBAL_LOAD_ASYNC_TO_LDS_B128 (ASYNCcnt),
//   double-buffered; last iteration peeled so the hot loop is branch-free.
// ---------------------------------------------------------------------------

namespace {
constexpr int kB   = 64;
constexpr int kS   = 128;
constexpr int kD   = 1024;
constexpr int kH   = 16;
constexpr int kHD  = 64;
constexpr int kL   = 6;
constexpr int kFFN = 4096;
constexpr int kM   = kB * kS;      // 8192 token rows
constexpr int kCH  = 2048;         // FFN row chunk (bounds scratch use)

constexpr int BM = 128, BN = 128, BK = 32;
constexpr int LDT = 40;            // ushort stride: 80B rows (16B aligned, bank-skewed)
}

typedef __bf16 bf16_t;
typedef bf16_t v16bf __attribute__((ext_vector_type(16)));
typedef float  v8f   __attribute__((ext_vector_type(8)));

__device__ __forceinline__ unsigned short f2bf(float f) {
  union { float f; unsigned u; } c; c.f = f;
  return (unsigned short)((c.u + 0x7FFFu + ((c.u >> 16) & 1u)) >> 16);
}
__device__ __forceinline__ void stval(float* p, float v) { *p = v; }
__device__ __forceinline__ void stval(bf16_t* p, float v) { *(unsigned short*)p = f2bf(v); }

// 16B global -> LDS async copy (CDNA5 async path, tracked by ASYNCcnt).
__device__ __forceinline__ void async_copy_b128(unsigned lds_addr, const void* gptr) {
  asm volatile("global_load_async_to_lds_b128 %0, %1, off"
               :: "v"(lds_addr), "v"(gptr)
               : "memory");
}

// ---------------------------------------------------------------------------
// fp32 -> bf16 bulk conversion (weights, once per layer). n multiple of 1024.
// ---------------------------------------------------------------------------
__global__ __launch_bounds__(256) void cvt_bf16_kernel(
    const float* __restrict__ src, bf16_t* __restrict__ dst) {
  size_t i = (size_t)blockIdx.x * 1024 + (size_t)threadIdx.x * 4;
  float4 v = *(const float4*)(src + i);
  unsigned p0 = (unsigned)f2bf(v.x) | ((unsigned)f2bf(v.y) << 16);
  unsigned p1 = (unsigned)f2bf(v.z) | ((unsigned)f2bf(v.w) << 16);
  unsigned* d = (unsigned*)((unsigned short*)dst + i);
  d[0] = p0; d[1] = p1;
}

// ---------------------------------------------------------------------------
// Fused GEMM: C[M,N] = epilogue( (A[M,K] @ W[N,K]^T + bias[N]) * alpha )
// A, W bf16; C fp32 or bf16 (OUTBF); optional ReLU / fp32 residual add.
// 256 threads = 8 wave32, 128x128 tile, K-step 32, double-buffered async LDS.
// Each wave: 2x4 WMMA 16x16 tiles (32x64 patch), 8 WMMAs per K-step.
// ---------------------------------------------------------------------------
template <bool RELU, bool RESID, bool OUTBF>
__global__ __launch_bounds__(256) void gemm_bf16_kernel(
    const bf16_t* __restrict__ A, const bf16_t* __restrict__ W,
    const float* __restrict__ bias, const float* __restrict__ resid,
    void* __restrict__ Cv, int N, int K, float alpha) {
  __shared__ unsigned short As[2][BM][LDT];
  __shared__ unsigned short Bs[2][BN][LDT];

  const int tid  = threadIdx.x;
  const int lane = tid & 31;
  const int wave = tid >> 5;
  const int hf   = lane >> 4;        // lane half (0/1)
  const int l16  = lane & 15;
  const int wm   = (wave & 3) * 32;  // wave M offset in tile
  const int wn   = (wave >> 2) * 64; // wave N offset in tile

  const int rowBase = blockIdx.y * BM;
  const int colBase = blockIdx.x * BN;

  // Each thread stages 2x16B of A and 2x16B of B per tile (4 async ops/thread).
  const int cr0 = tid >> 2;                 // rows 0..63
  const int cq0 = (tid & 3) * 8;            // bf16 col offset (8 elems = 16B)
  const int cr1 = cr0 + 64;                 // rows 64..127

  auto issue_tile = [&](int kt, int buf) {
    async_copy_b128((unsigned)(uintptr_t)&As[buf][cr0][cq0],
                    A + (size_t)(rowBase + cr0) * K + kt + cq0);
    async_copy_b128((unsigned)(uintptr_t)&Bs[buf][cr0][cq0],
                    W + (size_t)(colBase + cr0) * K + kt + cq0);
    async_copy_b128((unsigned)(uintptr_t)&As[buf][cr1][cq0],
                    A + (size_t)(rowBase + cr1) * K + kt + cq0);
    async_copy_b128((unsigned)(uintptr_t)&Bs[buf][cr1][cq0],
                    W + (size_t)(colBase + cr1) * K + kt + cq0);
  };

  v8f zero = {};
  v8f acc[2][4];
#pragma unroll
  for (int i = 0; i < 2; ++i)
#pragma unroll
    for (int j = 0; j < 4; ++j) acc[i][j] = zero;

  // Fragments per ISA 16-bit A(16x32)/B(32x16) wave32 lane layouts; the
  // per-lane 32-bit reads below are contiguous 16B runs -> ds_load_b128.
  auto compute_tile = [&](int cur) {
    union { v16bf v; unsigned u[8]; } afr[2], bfr[4];
#pragma unroll
    for (int i = 0; i < 2; ++i) {
      const unsigned short* row = &As[cur][wm + i * 16 + l16][0];
#pragma unroll
      for (int v = 0; v < 8; ++v) {
        int k = ((v < 4) ? (2 * v) : (16 + 2 * (v - 4))) + hf * 8;
        afr[i].u[v] = *(const unsigned*)(row + k);
      }
    }
#pragma unroll
    for (int j = 0; j < 4; ++j) {
      const unsigned short* row = &Bs[cur][wn + j * 16 + l16][0];
#pragma unroll
      for (int v = 0; v < 8; ++v) {
        int k = ((v < 4) ? (2 * v) : (16 + 2 * (v - 4))) + hf * 8;
        bfr[j].u[v] = *(const unsigned*)(row + k);
      }
    }
#pragma unroll
    for (int i = 0; i < 2; ++i)
#pragma unroll
      for (int j = 0; j < 4; ++j)
        acc[i][j] = __builtin_amdgcn_wmma_f32_16x16x32_bf16(
            false, afr[i].v, false, bfr[j].v, (short)0, acc[i][j], false, false);
  };

  const int nT = K / BK;              // >= 2 for all our shapes
  issue_tile(0, 0);

  // Steady state: branch-free body (issue next, wait current, compute).
  for (int t = 0; t < nT - 1; ++t) {
    const int cur = t & 1;
    issue_tile((t + 1) * BK, cur ^ 1);
    asm volatile("s_wait_asynccnt 0x4" ::: "memory");  // tile t landed
    __syncthreads();
    compute_tile(cur);
    __syncthreads();
  }
  // Peeled last tile.
  asm volatile("s_wait_asynccnt 0x0" ::: "memory");
  __syncthreads();
  compute_tile((nT - 1) & 1);

  // Epilogue: bias, alpha, relu, residual; C/D layout (VGPR r, lane halves).
#pragma unroll
  for (int i = 0; i < 2; ++i) {
#pragma unroll
    for (int j = 0; j < 4; ++j) {
      int col = colBase + wn + j * 16 + l16;
      float bv = bias[col];
#pragma unroll
      for (int r = 0; r < 8; ++r) {
        int row = rowBase + wm + i * 16 + hf * 8 + r;
        float val = (acc[i][j][r] + bv) * alpha;
        if (RELU)  val = val > 0.0f ? val : 0.0f;
        if (RESID) val += resid[(size_t)row * N + col];
        if constexpr (OUTBF) {
          bf16_t* C = (bf16_t*)Cv;
          stval(&C[(size_t)row * N + col], val);
        } else {
          float* C = (float*)Cv;
          C[(size_t)row * N + col] = val;
        }
      }
    }
  }
}

// ---------------------------------------------------------------------------
// Embedding + positional encoding: X[row] = emb[id]*sqrt(D) + pos[s]  (fp32)
// ---------------------------------------------------------------------------
__global__ __launch_bounds__(256) void embed_kernel(
    const int* __restrict__ ids, const float* __restrict__ emb,
    const float* __restrict__ pos, float* __restrict__ X) {
  int row = blockIdx.x;              // b*S + s
  int s   = row & (kS - 1);
  int tid = threadIdx.x;
  int id  = ids[row];
  const float sc = 32.0f;            // sqrt(1024)
#pragma unroll
  for (int i = 0; i < 4; ++i) {
    int c = tid + i * 256;
    X[(size_t)row * kD + c] = emb[(size_t)id * kD + c] * sc + pos[(size_t)s * kD + c];
  }
}

// ---------------------------------------------------------------------------
// LayerNorm over D=1024, one row per 256-thread block. OutT = float or bf16.
// ---------------------------------------------------------------------------
template <typename OutT>
__global__ __launch_bounds__(256) void ln_kernel(
    const float* __restrict__ X, const float* __restrict__ g,
    const float* __restrict__ b, OutT* __restrict__ Y) {
  __shared__ float red[256];
  const int row = blockIdx.x;
  const int tid = threadIdx.x;
  const float* xr = X + (size_t)row * kD;
  float v[4];
  float s = 0.0f;
#pragma unroll
  for (int i = 0; i < 4; ++i) { v[i] = xr[tid + i * 256]; s += v[i]; }
  red[tid] = s;
  __syncthreads();
  for (int o = 128; o > 0; o >>= 1) {
    if (tid < o) red[tid] += red[tid + o];
    __syncthreads();
  }
  float mean = red[0] * (1.0f / kD);
  __syncthreads();
  float s2 = 0.0f;
#pragma unroll
  for (int i = 0; i < 4; ++i) { float d = v[i] - mean; s2 += d * d; }
  red[tid] = s2;
  __syncthreads();
  for (int o = 128; o > 0; o >>= 1) {
    if (tid < o) red[tid] += red[tid + o];
    __syncthreads();
  }
  float rstd = rsqrtf(red[0] * (1.0f / kD) + 1e-5f);
  OutT* yr = Y + (size_t)row * kD;
#pragma unroll
  for (int i = 0; i < 4; ++i) {
    int c = tid + i * 256;
    stval(&yr[c], (v[i] - mean) * rstd * g[c] + b[c]);
  }
}

// ---------------------------------------------------------------------------
// Attention per (b,h): S=128 queries (one per thread), HD=64, fp32 math.
// Q carries 1/sqrt(HD). Pad mask == keys restricted to len[b]. bf16 output.
// ---------------------------------------------------------------------------
__global__ __launch_bounds__(128) void attn_kernel(
    const float* __restrict__ Q, const float* __restrict__ Kt,
    const float* __restrict__ Vt, const int* __restrict__ len,
    bf16_t* __restrict__ O) {
  __shared__ float ks[kS][kHD];
  __shared__ float vs[kS][kHD];
  int b = blockIdx.x, h = blockIdx.y;
  int tid = threadIdx.x;             // 128 threads
  size_t base = (size_t)b * kS * kD + (size_t)h * kHD;
  for (int i = 0; i < 64; ++i) {     // coalesced staging of K,V rows
    int e = i * 128 + tid;
    int r = e >> 6, c = e & 63;
    ks[r][c] = Kt[base + (size_t)r * kD + c];
    vs[r][c] = Vt[base + (size_t)r * kD + c];
  }
  float q[kHD];
#pragma unroll
  for (int d = 0; d < kHD; ++d) q[d] = Q[base + (size_t)tid * kD + d];
  __syncthreads();

  int n = len[b];                    // >= 1 per reference
  float m = -3.0e38f;
  for (int j = 0; j < n; ++j) {
    float sc = 0.0f;
#pragma unroll
    for (int d = 0; d < kHD; ++d) sc += q[d] * ks[j][d];
    m = fmaxf(m, sc);
  }
  float sum = 0.0f;
  float o[kHD];
#pragma unroll
  for (int d = 0; d < kHD; ++d) o[d] = 0.0f;
  for (int j = 0; j < n; ++j) {
    float sc = 0.0f;
#pragma unroll
    for (int d = 0; d < kHD; ++d) sc += q[d] * ks[j][d];
    float p = __expf(sc - m);
    sum += p;
#pragma unroll
    for (int d = 0; d < kHD; ++d) o[d] += p * vs[j][d];
  }
  float inv = 1.0f / sum;
#pragma unroll
  for (int d = 0; d < kHD; ++d)
    stval(&O[base + (size_t)tid * kD + d], o[d] * inv);
}

// ---------------------------------------------------------------------------
// Host-side orchestration.
// ---------------------------------------------------------------------------
extern "C" void kernel_launch(void* const* d_in, const int* in_sizes, int n_in,
                              void* d_out, int out_size, void* d_ws, size_t ws_size,
                              hipStream_t stream) {
  (void)in_sizes; (void)n_in; (void)out_size; (void)ws_size;

  const int*   input_ids = (const int*)d_in[0];
  const int*   input_len = (const int*)d_in[1];
  const float* embed_w   = (const float*)d_in[2];
  const float* pos       = (const float*)d_in[3];
  const float* Wq = (const float*)d_in[4];  const float* bq = (const float*)d_in[5];
  const float* Wk = (const float*)d_in[6];  const float* bk = (const float*)d_in[7];
  const float* Wv = (const float*)d_in[8];  const float* bv = (const float*)d_in[9];
  const float* Wo = (const float*)d_in[10]; const float* bo = (const float*)d_in[11];
  const float* ln1_g = (const float*)d_in[12]; const float* ln1_b = (const float*)d_in[13];
  const float* W1 = (const float*)d_in[14]; const float* b1 = (const float*)d_in[15];
  const float* W2 = (const float*)d_in[16]; const float* b2 = (const float*)d_in[17];
  const float* ln2_g = (const float*)d_in[18]; const float* ln2_b = (const float*)d_in[19];
  const float* lnf_g = (const float*)d_in[20]; const float* lnf_b = (const float*)d_in[21];

  // Workspace: X fp32 | Hb bf16(/attn-out) | Q,K,V fp32 | Rb bf16 | Wb bf16.
  char* w = (char*)d_ws;
  float*  X  = (float*)w;  w += (size_t)kM * kD * 4;
  bf16_t* Hb = (bf16_t*)w; w += (size_t)kM * kD * 2;
  float*  Qb = (float*)w;  w += (size_t)kM * kD * 4;
  float*  Kb = (float*)w;  w += (size_t)kM * kD * 4;
  float*  Vb = (float*)w;  w += (size_t)kM * kD * 4;
  bf16_t* Rb = (bf16_t*)w; w += (size_t)kCH * kFFN * 2;
  bf16_t* Wb = (bf16_t*)w;                    // per-layer bf16 weights (reused)
  bf16_t* wqB = Wb;
  bf16_t* wkB = wqB + (size_t)kD * kD;
  bf16_t* wvB = wkB + (size_t)kD * kD;
  bf16_t* woB = wvB + (size_t)kD * kD;
  bf16_t* w1B = woB + (size_t)kD * kD;
  bf16_t* w2B = w1B + (size_t)kFFN * kD;

  embed_kernel<<<kM, 256, 0, stream>>>(input_ids, embed_w, pos, X);

  const dim3 gD (kD / BN,   kM / BM);   // 8 x 64
  const dim3 gF1(kFFN / BN, kCH / BM);  // 32 x 16
  const dim3 gF2(kD / BN,   kCH / BM);  // 8 x 16
  const int  nDD = kD * kD / 1024;      // cvt blocks per DxD matrix
  const int  nDF = kD * kFFN / 1024;
  const float scaling = 0.125f;         // HD^-0.5

  for (int l = 0; l < kL; ++l) {
    // Convert this layer's weights to bf16 once (trivial vs GEMM time).
    cvt_bf16_kernel<<<nDD, 256, 0, stream>>>(Wq + (size_t)l * kD * kD, wqB);
    cvt_bf16_kernel<<<nDD, 256, 0, stream>>>(Wk + (size_t)l * kD * kD, wkB);
    cvt_bf16_kernel<<<nDD, 256, 0, stream>>>(Wv + (size_t)l * kD * kD, wvB);
    cvt_bf16_kernel<<<nDD, 256, 0, stream>>>(Wo + (size_t)l * kD * kD, woB);
    cvt_bf16_kernel<<<nDF, 256, 0, stream>>>(W1 + (size_t)l * kFFN * kD, w1B);
    cvt_bf16_kernel<<<nDF, 256, 0, stream>>>(W2 + (size_t)l * kD * kFFN, w2B);

    ln_kernel<bf16_t><<<kM, 256, 0, stream>>>(
        X, ln1_g + (size_t)l * kD, ln1_b + (size_t)l * kD, Hb);

    gemm_bf16_kernel<false, false, false><<<gD, 256, 0, stream>>>(
        Hb, wqB, bq + (size_t)l * kD, nullptr, Qb, kD, kD, scaling);
    gemm_bf16_kernel<false, false, false><<<gD, 256, 0, stream>>>(
        Hb, wkB, bk + (size_t)l * kD, nullptr, Kb, kD, kD, 1.0f);
    gemm_bf16_kernel<false, false, false><<<gD, 256, 0, stream>>>(
        Hb, wvB, bv + (size_t)l * kD, nullptr, Vb, kD, kD, 1.0f);

    attn_kernel<<<dim3(kB, kH), 128, 0, stream>>>(Qb, Kb, Vb, input_len, Hb);

    // x = x + attn @ Wo^T + bo   (residual fused)
    gemm_bf16_kernel<false, true, false><<<gD, 256, 0, stream>>>(
        Hb, woB, bo + (size_t)l * kD, X, X, kD, kD, 1.0f);

    ln_kernel<bf16_t><<<kM, 256, 0, stream>>>(
        X, ln2_g + (size_t)l * kD, ln2_b + (size_t)l * kD, Hb);

    // FFN, chunked over rows to bound the 4096-wide intermediate.
    for (int c = 0; c < kM / kCH; ++c) {
      const bf16_t* hA = Hb + (size_t)c * kCH * kD;
      float*        xC = X  + (size_t)c * kCH * kD;
      gemm_bf16_kernel<true, false, true><<<gF1, 256, 0, stream>>>(
          hA, w1B, b1 + (size_t)l * kFFN, nullptr, Rb, kFFN, kD, 1.0f);
      gemm_bf16_kernel<false, true, false><<<gF2, 256, 0, stream>>>(
          Rb, w2B, b2 + (size_t)l * kD, xC, xC, kD, kFFN, 1.0f);
    }
  }

  ln_kernel<float><<<kM, 256, 0, stream>>>(X, lnf_g, lnf_b, (float*)d_out);
}
